// Net_79568564126090
// MI455X (gfx1250) — compile-verified, matrix operand order checked by reading, not run on hardware
//
#include <hip/hip_runtime.h>
#include <hip/hip_bf16.h>

// ---------------- problem constants ----------------
#define N_NODES 50000
#define N_EDGES 400000
#define IN_CH   128
#define HID     1024
#define OUT_CH  47
#define K1      (2*IN_CH)   // 256  concat [mean|x]
#define K2      (2*HID)     // 2048 concat [mean2|h]
#define PADN2   48          // OUT_CH padded to 16
#define NKT1    (K1/32)     // 8
#define NKT2    (K2/32)     // 64
#define NNT1    (HID/16)    // 64
#define NNT2    (PADN2/16)  // 3
#define MT      (N_NODES/16) // 3125 row tiles

typedef __attribute__((ext_vector_type(16))) __bf16 v16bf;
typedef __attribute__((ext_vector_type(8)))  float  v8f;

// ---------------- device helpers ----------------
__device__ __forceinline__ int imax(int a, int b) { return a > b ? a : b; }

// A fragment: row-major bf16 row pointer already offset by (lane>=16 ? 8 : 0).
// Per ISA 16-bit A layout: v0..v3 hold k+0..7, v4..v7 hold k+16..23.
__device__ __forceinline__ v16bf load_frag_a(const __hip_bfloat16* p_) {
  const __bf16* p = reinterpret_cast<const __bf16*>(p_);
  union { v16bf v; int4 q[2]; } u;
  u.q[0] = *reinterpret_cast<const int4*>(p);
  u.q[1] = *reinterpret_cast<const int4*>(p + 16);
  return u.v;
}

// B fragment from pre-packed tiles: 16 contiguous bf16 per lane, already in
// VGPR order (elements 0..15 = v0.lo,v0.hi,...,v7.hi).
__device__ __forceinline__ v16bf load_frag_packed(const __hip_bfloat16* p_) {
  const __bf16* p = reinterpret_cast<const __bf16*>(p_);
  union { v16bf v; int4 q[2]; } u;
  u.q[0] = *reinterpret_cast<const int4*>(p);
  u.q[1] = *reinterpret_cast<const int4*>(p + 8);
  return u.v;
}

__device__ __forceinline__ v8f wmma_bf16(v16bf a, v16bf b, v8f c) {
  return __builtin_amdgcn_wmma_f32_16x16x32_bf16(
      /*neg_a=*/false, a, /*neg_b=*/false, b,
      /*c_mod=*/(short)0, c, /*reuse_a=*/false, /*reuse_b=*/false);
}

// ---------------- graph prep kernels ----------------
__global__ void k_zero_i32(int* p, int n) {
  int i = blockIdx.x * blockDim.x + threadIdx.x;
  if (i < n) p[i] = 0;
}

__global__ void k_count_deg(const int* dst, int* cnt, int e) {
  int i = blockIdx.x * blockDim.x + threadIdx.x;
  if (i < e) atomicAdd(&cnt[dst[i]], 1);
}

// single-workgroup exclusive scan over cnt -> rowptr (n+1) and cursor (n)
__global__ void k_scan(const int* cnt, int* rowptr, int* cursor, int n) {
  __shared__ int sh[1024];
  __shared__ int carry_s;
  int tid = threadIdx.x;
  if (tid == 0) carry_s = 0;
  __syncthreads();
  for (int base = 0; base < n; base += 1024) {
    int i = base + tid;
    int v = (i < n) ? cnt[i] : 0;
    sh[tid] = v;
    __syncthreads();
    for (int off = 1; off < 1024; off <<= 1) {
      int t = (tid >= off) ? sh[tid - off] : 0;
      __syncthreads();
      sh[tid] += t;
      __syncthreads();
    }
    int excl = sh[tid] - v;
    int carry = carry_s;
    if (i < n) { rowptr[i] = carry + excl; cursor[i] = carry + excl; }
    __syncthreads();
    if (tid == 0) carry_s = carry + sh[1023];
    __syncthreads();
  }
  if (tid == 0) rowptr[n] = carry_s;
}

__global__ void k_fill_csr(const int* src, const int* dst, int* cursor,
                           int* csr_src, int e) {
  int i = blockIdx.x * blockDim.x + threadIdx.x;
  if (i < e) {
    int p = atomicAdd(&cursor[dst[i]], 1);
    csr_src[p] = src[i];
  }
}

// ---------------- weight packing ----------------
// Pack W = [srcA; srcB] (rows 0..splitK-1 from srcA, rest srcB), row-major
// K x Nw fp32, into WMMA-B tile order: tile (nt,kt) at (nt*numKt+kt)*512,
// within tile lane*16 + e, e -> (v,pair) per ISA 16-bit B layout.
__global__ void k_pack_w(const float* srcA, const float* srcB, int splitK,
                         int K, int Nw, int numKt, int numNt,
                         __hip_bfloat16* dst) {
  int idx = blockIdx.x * blockDim.x + threadIdx.x;
  int total = numKt * numNt * 512;
  if (idx >= total) return;
  int e  = idx & 15;
  int l  = (idx >> 4) & 31;
  int t  = idx >> 9;
  int kt = t % numKt;
  int nt = t / numKt;
  int koff = (e < 8 ? e : e + 8) + ((l >= 16) ? 8 : 0);
  int k = kt * 32 + koff;
  int n = nt * 16 + (l & 15);
  float v = 0.f;
  if (n < Nw) v = (k < splitK) ? srcA[k * Nw + n] : srcB[(k - splitK) * Nw + n];
  dst[idx] = __float2bfloat16(v);
}

// ---------------- layer 1: aggregate + build A1 = [mean|x] bf16 ----------------
__global__ void k_agg1(const float* x, const int* rowptr, const int* csr_src,
                       __hip_bfloat16* A1) {
  int node = blockIdx.x;
  int c = threadIdx.x;            // 128 threads = channels
  int beg = rowptr[node], end = rowptr[node + 1];
  float s = 0.f;
  for (int j = beg; j < end; ++j) s += x[csr_src[j] * IN_CH + c];
  float mean = s / (float)imax(end - beg, 1);
  A1[(size_t)node * K1 + c]         = __float2bfloat16(mean);
  A1[(size_t)node * K1 + IN_CH + c] = __float2bfloat16(x[node * IN_CH + c]);
}

// ---------------- GEMM1: h = relu(A1 @ W1cat + b1), bf16 out ----------------
__global__ void k_gemm1(const __hip_bfloat16* A1, const __hip_bfloat16* W1p,
                        const float* b1, __hip_bfloat16* h) {
  int lane = threadIdx.x & 31;
  int wave = (blockIdx.x * blockDim.x + threadIdx.x) >> 5;
  int mt  = wave >> 4;            // 0..3124
  int ntg = wave & 15;            // covers n-tiles ntg*4 .. ntg*4+3
  int m   = mt * 16 + (lane & 15);
  int hi8 = (lane >> 4) * 8;
  const __hip_bfloat16* arow = A1 + (size_t)m * K1 + hi8;
  const __hip_bfloat16* bbase =
      W1p + (size_t)(ntg * 4) * NKT1 * 512 + lane * 16;
  v8f a0 = {}, a1 = {}, a2 = {}, a3 = {};
  // software pipeline: all loads of iteration kt (and next A frag) issue
  // before the WMMAs of iteration kt consume them.
  v16bf af = load_frag_a(arow);
#pragma unroll
  for (int kt = 0; kt < NKT1; ++kt) {
    const __hip_bfloat16* bp = bbase + (size_t)kt * 512;
    v16bf bf0 = load_frag_packed(bp);
    v16bf bf1 = load_frag_packed(bp + 1 * NKT1 * 512);
    v16bf bf2 = load_frag_packed(bp + 2 * NKT1 * 512);
    v16bf bf3 = load_frag_packed(bp + 3 * NKT1 * 512);
    v16bf af_next = (kt + 1 < NKT1) ? load_frag_a(arow + (kt + 1) * 32) : af;
    a0 = wmma_bf16(af, bf0, a0);
    a1 = wmma_bf16(af, bf1, a1);
    a2 = wmma_bf16(af, bf2, a2);
    a3 = wmma_bf16(af, bf3, a3);
    af = af_next;
  }
  int mb = mt * 16 + (lane >> 4) * 8;
  int nl = lane & 15;
  v8f accs[4] = {a0, a1, a2, a3};
#pragma unroll
  for (int tt = 0; tt < 4; ++tt) {
    int n = (ntg * 4 + tt) * 16 + nl;
    float bias = b1[n];
#pragma unroll
    for (int j = 0; j < 8; ++j) {
      float v = accs[tt][j] + bias;
      v = v > 0.f ? v : 0.f;
      h[(size_t)(mb + j) * HID + n] = __float2bfloat16(v);
    }
  }
}

// ---------------- layer 2 aggregation: mean2 bf16 ----------------
__global__ void k_agg2(const __hip_bfloat16* h, const int* rowptr,
                       const int* csr_src, __hip_bfloat16* mean2) {
  int node = blockIdx.x;
  int c = threadIdx.x * 4;        // 256 threads x 4 channels
  int beg = rowptr[node], end = rowptr[node + 1];
  float s0 = 0.f, s1 = 0.f, s2 = 0.f, s3 = 0.f;
  for (int j = beg; j < end; ++j) {
    const __hip_bfloat16* hp = h + (size_t)csr_src[j] * HID + c;
    s0 += __bfloat162float(hp[0]);
    s1 += __bfloat162float(hp[1]);
    s2 += __bfloat162float(hp[2]);
    s3 += __bfloat162float(hp[3]);
  }
  float inv = 1.f / (float)imax(end - beg, 1);
  __hip_bfloat16* mp = mean2 + (size_t)node * HID + c;
  mp[0] = __float2bfloat16(s0 * inv);
  mp[1] = __float2bfloat16(s1 * inv);
  mp[2] = __float2bfloat16(s2 * inv);
  mp[3] = __float2bfloat16(s3 * inv);
}

// ---------------- GEMM2: out = [mean2|h] @ W2cat + b2 (fp32) ----------------
__global__ void k_gemm2(const __hip_bfloat16* mean2, const __hip_bfloat16* h,
                        const __hip_bfloat16* W2p, const float* b2,
                        float* out) {
  int lane = threadIdx.x & 31;
  int wave = (blockIdx.x * blockDim.x + threadIdx.x) >> 5;
  if (wave >= MT) return;
  int m   = wave * 16 + (lane & 15);
  int hi8 = (lane >> 4) * 8;
  const __hip_bfloat16* arow_m = mean2 + (size_t)m * HID + hi8; // kt 0..31
  const __hip_bfloat16* arow_h = h + (size_t)m * HID + hi8;     // kt 32..63
  const __hip_bfloat16* bbase = W2p + lane * 16;
  v8f a0 = {}, a1 = {}, a2 = {};
  v16bf af = load_frag_a(arow_m);
  // unroll x2 over kt; prefetch next A fragment across the WMMA group
#pragma unroll 2
  for (int kt = 0; kt < NKT2; ++kt) {
    const __hip_bfloat16* bp = bbase + (size_t)kt * 512;
    v16bf bf0 = load_frag_packed(bp);
    v16bf bf1 = load_frag_packed(bp + 1 * NKT2 * 512);
    v16bf bf2 = load_frag_packed(bp + 2 * NKT2 * 512);
    int ktn = kt + 1;
    v16bf af_next = af;
    if (ktn < NKT2) {
      const __hip_bfloat16* an =
          (ktn < NKT2 / 2) ? (arow_m + ktn * 32) : (arow_h + (ktn - NKT2 / 2) * 32);
      af_next = load_frag_a(an);
    }
    a0 = wmma_bf16(af, bf0, a0);
    a1 = wmma_bf16(af, bf1, a1);
    a2 = wmma_bf16(af, bf2, a2);
    af = af_next;
  }
  int mb = wave * 16 + (lane >> 4) * 8;
  int nl = lane & 15;
  v8f accs[3] = {a0, a1, a2};
#pragma unroll
  for (int tt = 0; tt < 3; ++tt) {
    int n = tt * 16 + nl;
    if (n < OUT_CH) {
      float bias = b2[n];
#pragma unroll
      for (int j = 0; j < 8; ++j)
        out[(size_t)(mb + j) * OUT_CH + n] = accs[tt][j] + bias;
    }
  }
}

// ---------------- log_softmax, one wave per node, in place ----------------
__global__ void k_logsoftmax(float* out, int n_nodes) {
  int lane = threadIdx.x & 31;
  int wave = (blockIdx.x * blockDim.x + threadIdx.x) >> 5;
  if (wave >= n_nodes) return;
  float* row = out + (size_t)wave * OUT_CH;
  float v0 = (lane < OUT_CH) ? row[lane] : -3.402823466e38f;
  float v1 = (lane + 32 < OUT_CH) ? row[lane + 32] : -3.402823466e38f;
  float mx = fmaxf(v0, v1);
#pragma unroll
  for (int o = 16; o > 0; o >>= 1) mx = fmaxf(mx, __shfl_xor(mx, o, 32));
  float s = (lane < OUT_CH ? expf(v0 - mx) : 0.f) +
            (lane + 32 < OUT_CH ? expf(v1 - mx) : 0.f);
#pragma unroll
  for (int o = 16; o > 0; o >>= 1) s += __shfl_xor(s, o, 32);
  float lse = mx + logf(s);
  if (lane < OUT_CH) row[lane] = v0 - lse;
  if (lane + 32 < OUT_CH) row[lane + 32] = v1 - lse;
}

// ---------------- launcher ----------------
extern "C" void kernel_launch(void* const* d_in, const int* in_sizes, int n_in,
                              void* d_out, int out_size, void* d_ws, size_t ws_size,
                              hipStream_t stream) {
  const float* x   = (const float*)d_in[0];
  const int*   ei  = (const int*)d_in[1];     // [2, E]
  const float* Wl1 = (const float*)d_in[2];
  const float* Wr1 = (const float*)d_in[3];
  const float* b1  = (const float*)d_in[4];
  const float* Wl2 = (const float*)d_in[5];
  const float* Wr2 = (const float*)d_in[6];
  const float* b2  = (const float*)d_in[7];
  float* out = (float*)d_out;

  const int* src = ei;
  const int* dst = ei + N_EDGES;

  // carve workspace
  char* ws = (char*)d_ws;
  auto carve = [&](size_t bytes) -> char* {
    char* p = ws;
    ws += (bytes + 255) & ~(size_t)255;
    return p;
  };
  int* cnt      = (int*)carve((size_t)N_NODES * 4);
  int* rowptr   = (int*)carve((size_t)(N_NODES + 1) * 4);
  int* cursor   = (int*)carve((size_t)N_NODES * 4);
  int* csr_src  = (int*)carve((size_t)N_EDGES * 4);
  __hip_bfloat16* A1    = (__hip_bfloat16*)carve((size_t)N_NODES * K1 * 2);
  __hip_bfloat16* W1p   = (__hip_bfloat16*)carve((size_t)NKT1 * NNT1 * 512 * 2);
  __hip_bfloat16* W2p   = (__hip_bfloat16*)carve((size_t)NKT2 * NNT2 * 512 * 2);
  __hip_bfloat16* hbuf  = (__hip_bfloat16*)carve((size_t)N_NODES * HID * 2);
  __hip_bfloat16* mean2 = (__hip_bfloat16*)carve((size_t)N_NODES * HID * 2);

  // 1. degree count + CSR build
  k_zero_i32<<<(N_NODES + 255) / 256, 256, 0, stream>>>(cnt, N_NODES);
  k_count_deg<<<(N_EDGES + 255) / 256, 256, 0, stream>>>(dst, cnt, N_EDGES);
  k_scan<<<1, 1024, 0, stream>>>(cnt, rowptr, cursor, N_NODES);
  k_fill_csr<<<(N_EDGES + 255) / 256, 256, 0, stream>>>(src, dst, cursor,
                                                        csr_src, N_EDGES);
  // 2. pack weights into WMMA-B tile layout (bf16)
  {
    int tot1 = NKT1 * NNT1 * 512;
    k_pack_w<<<(tot1 + 255) / 256, 256, 0, stream>>>(Wl1, Wr1, IN_CH, K1, HID,
                                                     NKT1, NNT1, W1p);
    int tot2 = NKT2 * NNT2 * 512;
    k_pack_w<<<(tot2 + 255) / 256, 256, 0, stream>>>(Wl2, Wr2, HID, K2, OUT_CH,
                                                     NKT2, NNT2, W2p);
  }
  // 3. layer 1
  k_agg1<<<N_NODES, IN_CH, 0, stream>>>(x, rowptr, csr_src, A1);
  k_gemm1<<<(MT * 16 * 32) / 256, 256, 0, stream>>>(A1, W1p, b1, hbuf);
  // 4. layer 2
  k_agg2<<<N_NODES, 256, 0, stream>>>(hbuf, rowptr, csr_src, mean2);
  k_gemm2<<<(MT * 32 + 255) / 256, 256, 0, stream>>>(mean2, hbuf, W2p, b2, out);
  // 5. log_softmax in place
  k_logsoftmax<<<(N_NODES * 32 + 255) / 256, 256, 0, stream>>>(out, N_NODES);
}
